// SwitchAttention_49177375539842
// MI455X (gfx1250) — matrix operand order repaired and checked by reading, not compiled, not run on hardware
//
#include <hip/hip_runtime.h>
#include <hip/hip_bf16.h>
#include <stdint.h>

// Switch-MoE LoRA attention forward for MI455X (gfx1250):
// bf16 WMMA everywhere + TDM (tensor_load_to_lds) tile staging.
#define H_   768
#define NH_  12
#define DH_  64
#define E_   2
#define R_   128
#define B_   32
#define S_   512
#define SCALE_ 0.125f   // 1/sqrt(64)

typedef __attribute__((ext_vector_type(16))) __bf16 v16bf;
typedef __attribute__((ext_vector_type(8)))  float  v8f;
typedef unsigned int u32x4 __attribute__((ext_vector_type(4)));
typedef int          i32x8 __attribute__((ext_vector_type(8)));
typedef int          i32x4 __attribute__((ext_vector_type(4)));

union Frag { v16bf v; uint4 q[2]; };

__device__ __forceinline__ uint16_t f2bf(float f) {
  union { float f; uint32_t u; } cv; cv.f = f;
  uint32_t u = cv.u;
  return (uint16_t)((u + 0x7FFFu + ((u >> 16) & 1u)) >> 16);  // RNE
}

__device__ __forceinline__ v8f wmma_bf16(v16bf a, v16bf b, v8f c) {
  return __builtin_amdgcn_wmma_f32_16x16x32_bf16(false, a, false, b, (short)0, c, false, false);
}

// ---- TDM: DMA a 2-D bf16 tile [height x width] (row stride in elements) into LDS.
// D# layout per CDNA5 ISA; 6-arg builtin on this toolchain.
__device__ __forceinline__ void tdm_load_2d_to_lds(uint32_t lds_off, const void* gptr,
                                                   uint32_t width, uint32_t height,
                                                   uint32_t stride_elems) {
  uint64_t ga = (uint64_t)(uintptr_t)gptr;
  u32x4 g0;
  g0[0] = 1u;                                                 // count=1 (user D#)
  g0[1] = lds_off;                                            // lds_addr [63:32]
  g0[2] = (uint32_t)ga;                                       // global_addr lo
  g0[3] = (uint32_t)((ga >> 32) & 0x01FFFFFFu) | (2u << 30);  // addr hi | type=2
  i32x8 g1;
  g1[0] = (int)(1u << 16);                                    // data_size=1 -> 2 bytes
  g1[1] = (int)((width & 0xFFFFu) << 16);                     // tensor_dim0 lo16
  g1[2] = (int)(((width >> 16) & 0xFFFFu) |                   // tensor_dim0 hi16
                ((height & 0xFFFFu) << 16));                  // tensor_dim1 lo16
  g1[3] = (int)(((height >> 16) & 0xFFFFu) |                  // tensor_dim1 hi16
                ((width & 0xFFFFu) << 16));                   // tile_dim0
  g1[4] = (int)(height & 0xFFFFu);                            // tile_dim1 (tile_dim2=0)
  g1[5] = (int)stride_elems;                                  // tensor_dim0_stride lo32
  g1[6] = 0;
  g1[7] = 0;
  i32x4 gz4 = {0, 0, 0, 0};                                   // groups 2/3: unused dims
  i32x8 gz8 = {0, 0, 0, 0, 0, 0, 0, 0};
  __builtin_amdgcn_tensor_load_to_lds(g0, g1, gz4, gz4, gz8, 0);
}

// A-frag (16x32, M=row0.., K=k0..) from row-major bf16 [rows][ld] (global or LDS)
__device__ __forceinline__ v16bf load_a(const uint16_t* base, int ld, int row0, int k0, int lane) {
  int half = lane >> 4, l16 = lane & 15;
  const uint16_t* p = base + (size_t)(row0 + l16) * ld + k0 + 8 * half;
  Frag f;
  f.q[0] = *(const uint4*)p;        // K = k0 + 8*half + (0..7)
  f.q[1] = *(const uint4*)(p + 16); // K = k0 + 16 + 8*half + (0..7)
  return f.v;
}

// B-frag (32x16, K=k0..k0+31, N=n0..n0+15) from "N-major" bf16 [N][ld] (K contiguous)
__device__ __forceinline__ v16bf load_b(const uint16_t* base, int ld, int n0, int k0, int lane) {
  int half = lane >> 4, l16 = lane & 15;
  const uint16_t* p = base + (size_t)(n0 + l16) * ld + k0 + 8 * half;
  Frag f;
  f.q[0] = *(const uint4*)p;        // K = k0 + 8*half + (0..7)
  f.q[1] = *(const uint4*)(p + 16); // K = k0 + 16 + 8*half + (0..7)
  return f.v;
}

// ---------------- router: mean-pool -> 2 logits -> argmax ----------------
__global__ void router_kernel(const float* __restrict__ x, const float* __restrict__ sw,
                              const float* __restrict__ sb, int* __restrict__ routes) {
  __shared__ float pooled[H_];
  __shared__ float r0[256], r1[256];
  int b = blockIdx.x, tid = threadIdx.x;
  for (int h = tid; h < H_; h += 256) {
    float s = 0.f;
    const float* p = x + (size_t)b * S_ * H_ + h;
    for (int si = 0; si < S_; ++si) s += p[(size_t)si * H_];
    pooled[h] = s * (1.0f / S_);
  }
  __syncthreads();
  float p0 = 0.f, p1 = 0.f;
  for (int h = tid; h < H_; h += 256) {
    float pv = pooled[h];
    p0 += pv * sw[h * E_ + 0];
    p1 += pv * sw[h * E_ + 1];
  }
  r0[tid] = p0; r1[tid] = p1;
  __syncthreads();
  for (int off = 128; off > 0; off >>= 1) {
    if (tid < off) { r0[tid] += r0[tid + off]; r1[tid] += r1[tid + off]; }
    __syncthreads();
  }
  if (tid == 0) {
    float l0 = r0[0] + sb[0], l1 = r1[0] + sb[1];
    routes[b] = (l1 > l0) ? 1 : 0;  // argmax, first index wins ties
  }
}

// ---------------- f32 -> bf16 convert ----------------
__global__ void cvt_bf16_kernel(const float* __restrict__ in, uint16_t* __restrict__ out, int n) {
  int i = blockIdx.x * 256 + threadIdx.x;
  if (i < n) out[i] = f2bf(in[i]);
}

// ---------------- transpose+convert W[K][N] -> Wt[N][K] bf16 ----------------
__global__ void tconv_kernel(const float* __restrict__ W, uint16_t* __restrict__ Wt) {
  int id = blockIdx.x * 256 + threadIdx.x;
  if (id >= H_ * H_) return;
  int n = id / H_, k = id % H_;
  Wt[id] = f2bf(W[k * H_ + n]);
}

// ------------- fold LoRA: Wt[n][k] = bf16( w[k][n] + sum_r a[k][r] g[r][n] ) -------------
__global__ void lora_combine_kernel(const float* __restrict__ W, const float* __restrict__ A,
                                    const float* __restrict__ G, uint16_t* __restrict__ Wt) {
  int id = blockIdx.x * 256 + threadIdx.x;
  if (id >= H_ * H_) return;
  int n = id / H_, k = id % H_;
  float v = W[k * H_ + n];
  const float* a = A + (size_t)k * R_;
  const float* g = G + n;
#pragma unroll 4
  for (int r = 0; r < R_; ++r) v += a[r] * g[(size_t)r * H_];
  Wt[id] = f2bf(v);
}

// ------------- projection GEMM: out[b,s,:] = xb[b,s,:] @ Wt^T + bias -------------
// 32-row A-tile staged in LDS via TDM; each wave: 32(M) x 64(N), 8 WMMAs / 4 B-frags.
__global__ void __launch_bounds__(128)
proj_kernel(const uint16_t* __restrict__ Xb,
            const uint16_t* __restrict__ Wt0, const uint16_t* __restrict__ Wt1,
            const float* __restrict__ b0, const float* __restrict__ b1,
            const int* __restrict__ routes, int use_routes,
            uint16_t* __restrict__ out, int transposed) {
  __shared__ __align__(16) uint16_t smX[32 * H_];   // 48 KB
  int b = blockIdx.z;
  const uint16_t* Wt = Wt0; const float* bias = b0;
  if (use_routes) { int e = routes[b]; if (e) { Wt = Wt1; bias = b1; } }
  int lane = threadIdx.x & 31, wave = threadIdx.x >> 5;
  int half = lane >> 4, l16 = lane & 15;
  int row0 = blockIdx.x * 32;              // token tile within batch
  int n0 = (blockIdx.y * 4 + wave) * 64;   // 64 output features per wave

  if (threadIdx.x < 32) {                  // wave 0 drives the TDM
    tdm_load_2d_to_lds((uint32_t)(uintptr_t)smX,
                       Xb + ((size_t)b * S_ + row0) * H_, H_, 32, H_);
    __builtin_amdgcn_s_wait_tensorcnt(0);
  }
  __syncthreads();

  v8f accL[4] = {}, accH[4] = {};
  for (int k0 = 0; k0 < H_; k0 += 32) {
    v16bf a0 = load_a(smX, H_, 0, k0, lane);   // rows row0..row0+15
    v16bf a1 = load_a(smX, H_, 16, k0, lane);  // rows row0+16..row0+31
    if (k0 + 32 < H_)
      __builtin_prefetch(Wt + (size_t)(n0 + l16) * H_ + k0 + 32, 0, 1);
#pragma unroll
    for (int t = 0; t < 4; ++t) {
      v16bf bm = load_b(Wt, H_, n0 + 16 * t, k0, lane);
      accL[t] = wmma_bf16(a0, bm, accL[t]);
      accH[t] = wmma_bf16(a1, bm, accH[t]);
    }
  }
#pragma unroll
  for (int t = 0; t < 4; ++t) {
    int n = n0 + 16 * t + l16;
    float bv = bias[n];
    if (transposed) {
      union { uint16_t u[8]; uint4 q; } pk;
#pragma unroll
      for (int r = 0; r < 8; ++r) pk.u[r] = f2bf(accL[t][r] + bv);
      uint16_t* dst = out + ((size_t)b * H_ + n) * S_ + row0 + 8 * half;
      *(uint4*)dst = pk.q;
#pragma unroll
      for (int r = 0; r < 8; ++r) pk.u[r] = f2bf(accH[t][r] + bv);
      *(uint4*)(dst + 16) = pk.q;
    } else {
#pragma unroll
      for (int r = 0; r < 8; ++r)
        out[(size_t)(b * S_ + row0 + r + 8 * half) * H_ + n] = f2bf(accL[t][r] + bv);
#pragma unroll
      for (int r = 0; r < 8; ++r)
        out[(size_t)(b * S_ + row0 + 16 + r + 8 * half) * H_ + n] = f2bf(accH[t][r] + bv);
    }
  }
}

// ------------- flash-style attention: K and V^T heads staged in LDS via TDM -------------
__global__ void __launch_bounds__(128)
attn_kernel(const uint16_t* __restrict__ Q, const uint16_t* __restrict__ K,
            const uint16_t* __restrict__ Vt, const float* __restrict__ mask,
            uint16_t* __restrict__ Aout) {
  __shared__ __align__(16) uint16_t smK[S_ * DH_];     // 64 KB, [512][64]
  __shared__ __align__(16) uint16_t smV[DH_ * S_];     // 64 KB, [64][512]
  __shared__ __align__(16) uint16_t psm[4 * 16 * 32];  // 4 KB P staging
  int b = blockIdx.z, h = blockIdx.y;
  int lane = threadIdx.x & 31, wave = threadIdx.x >> 5;
  int half = lane >> 4, l16 = lane & 15;
  int q0 = blockIdx.x * 64 + wave * 16;
  uint16_t* sm = psm + wave * 512;

  const uint16_t* Qb = Q + (size_t)b * S_ * H_ + h * DH_;
  const uint16_t* Kb = K + (size_t)b * S_ * H_ + h * DH_;
  const uint16_t* Vb = Vt + ((size_t)b * H_ + h * DH_) * S_;  // [feat][s]
  const float* mk = mask + (size_t)b * S_;

  if (threadIdx.x < 32) {  // wave 0: DMA whole K head and V^T head into LDS
    tdm_load_2d_to_lds((uint32_t)(uintptr_t)smK, Kb, DH_, S_, H_);
    tdm_load_2d_to_lds((uint32_t)(uintptr_t)smV, Vb, S_, DH_, S_);
    __builtin_amdgcn_s_wait_tensorcnt(0);
  }

  v16bf aq0 = load_a(Qb, H_, q0, 0, lane);   // Q stays in registers
  v16bf aq1 = load_a(Qb, H_, q0, 32, lane);
  __syncthreads();

  v8f o[4] = {};
  float rmax[8], rsum[8];
#pragma unroll
  for (int r = 0; r < 8; ++r) { rmax[r] = -1e30f; rsum[r] = 0.f; }

  for (int kb = 0; kb < S_; kb += 32) {
    v8f s0 = {}, s1 = {};
    s0 = wmma_bf16(aq0, load_b(smK, DH_, kb, 0, lane), s0);
    s0 = wmma_bf16(aq1, load_b(smK, DH_, kb, 32, lane), s0);
    s1 = wmma_bf16(aq0, load_b(smK, DH_, kb + 16, 0, lane), s1);
    s1 = wmma_bf16(aq1, load_b(smK, DH_, kb + 16, 32, lane), s1);

    float m0add = (1.0f - mk[kb + l16]) * -10000.0f;
    float m1add = (1.0f - mk[kb + 16 + l16]) * -10000.0f;

    float alpha[8];
#pragma unroll
    for (int r = 0; r < 8; ++r) {
      float v0 = s0[r] * SCALE_ + m0add;
      float v1 = s1[r] * SCALE_ + m1add;
      float mx = fmaxf(v0, v1);
#pragma unroll
      for (int off = 1; off < 16; off <<= 1) mx = fmaxf(mx, __shfl_xor(mx, off));
      float mnew = fmaxf(rmax[r], mx);
      float al = __expf(rmax[r] - mnew);
      rmax[r] = mnew;
      float p0 = __expf(v0 - mnew), p1 = __expf(v1 - mnew);
      float rs = p0 + p1;
#pragma unroll
      for (int off = 1; off < 16; off <<= 1) rs += __shfl_xor(rs, off);
      rsum[r] = rsum[r] * al + rs;
      alpha[r] = al;
      int row = r + 8 * half;
      sm[row * 32 + l16] = f2bf(p0);        // same-wave LDS ops stay ordered
      sm[row * 32 + 16 + l16] = f2bf(p1);
    }
#pragma unroll
    for (int t = 0; t < 4; ++t) {
#pragma unroll
      for (int r = 0; r < 8; ++r) o[t][r] *= alpha[r];
    }
    // re-swizzle P (C-layout) -> A-fragment layout via LDS
    Frag pf;
    const uint16_t* pp = sm + l16 * 32 + 8 * half;
    pf.q[0] = *(const uint4*)pp;
    pf.q[1] = *(const uint4*)(pp + 16);
#pragma unroll
    for (int t = 0; t < 4; ++t) {
      v16bf bv = load_b(smV, S_, 16 * t, kb, lane);  // V^T from LDS
      o[t] = wmma_bf16(pf.v, bv, o[t]);
    }
  }
#pragma unroll
  for (int t = 0; t < 4; ++t) {
#pragma unroll
    for (int r = 0; r < 8; ++r) {
      float v = o[t][r] / rsum[r];
      Aout[(size_t)(b * S_ + q0 + r + 8 * half) * H_ + h * DH_ + 16 * t + l16] = f2bf(v);
    }
  }
}

// ------------- fused output projection + routed/common combine -------------
// 32-row tiles of both attention outputs staged via TDM; 16 WMMAs / 8 B-frags per k-step.
__global__ void __launch_bounds__(128)
outproj_kernel(const uint16_t* __restrict__ Ar, const uint16_t* __restrict__ Ac,
               const uint16_t* __restrict__ WoR0, const uint16_t* __restrict__ WoR1,
               const uint16_t* __restrict__ WoC,
               const float* __restrict__ boR0, const float* __restrict__ boR1,
               const float* __restrict__ boC,
               const int* __restrict__ routes, float* __restrict__ out) {
  __shared__ __align__(16) uint16_t smR[32 * H_];   // 48 KB
  __shared__ __align__(16) uint16_t smC[32 * H_];   // 48 KB
  int b = blockIdx.z;
  int e = routes[b];
  const uint16_t* WoR = e ? WoR1 : WoR0;
  const float* boR = e ? boR1 : boR0;
  int lane = threadIdx.x & 31, wave = threadIdx.x >> 5;
  int half = lane >> 4, l16 = lane & 15;
  int row0 = blockIdx.x * 32;
  int n0 = (blockIdx.y * 4 + wave) * 64;

  if (threadIdx.x < 32) {
    tdm_load_2d_to_lds((uint32_t)(uintptr_t)smR,
                       Ar + ((size_t)b * S_ + row0) * H_, H_, 32, H_);
    tdm_load_2d_to_lds((uint32_t)(uintptr_t)smC,
                       Ac + ((size_t)b * S_ + row0) * H_, H_, 32, H_);
    __builtin_amdgcn_s_wait_tensorcnt(0);
  }
  __syncthreads();

  v8f accL[4] = {}, accH[4] = {};
  for (int k0 = 0; k0 < H_; k0 += 32) {
    v16bf ar0 = load_a(smR, H_, 0, k0, lane);
    v16bf ar1 = load_a(smR, H_, 16, k0, lane);
    v16bf ac0 = load_a(smC, H_, 0, k0, lane);
    v16bf ac1 = load_a(smC, H_, 16, k0, lane);
#pragma unroll
    for (int t = 0; t < 4; ++t) {
      v16bf br = load_b(WoR, H_, n0 + 16 * t, k0, lane);
      accL[t] = wmma_bf16(ar0, br, accL[t]);
      accH[t] = wmma_bf16(ar1, br, accH[t]);
      v16bf bc = load_b(WoC, H_, n0 + 16 * t, k0, lane);
      accL[t] = wmma_bf16(ac0, bc, accL[t]);
      accH[t] = wmma_bf16(ac1, bc, accH[t]);
    }
  }
#pragma unroll
  for (int t = 0; t < 4; ++t) {
    int n = n0 + 16 * t + l16;
    float bv = boR[n] + boC[n];
#pragma unroll
    for (int r = 0; r < 8; ++r)
      out[(size_t)(b * S_ + row0 + r + 8 * half) * H_ + n] = accL[t][r] + bv;
#pragma unroll
    for (int r = 0; r < 8; ++r)
      out[(size_t)(b * S_ + row0 + 16 + r + 8 * half) * H_ + n] = accH[t][r] + bv;
  }
}

extern "C" void kernel_launch(void* const* d_in, const int* in_sizes, int n_in,
                              void* d_out, int out_size, void* d_ws, size_t ws_size,
                              hipStream_t stream) {
  const float* x        = (const float*)d_in[0];
  const float* switch_w = (const float*)d_in[1];
  const float* switch_b = (const float*)d_in[2];
  const float* c_wq = (const float*)d_in[3];
  const float* c_bq = (const float*)d_in[4];
  const float* c_wk = (const float*)d_in[5];
  const float* c_bk = (const float*)d_in[6];
  const float* c_wv = (const float*)d_in[7];
  const float* c_bv = (const float*)d_in[8];
  const float* c_wo = (const float*)d_in[9];
  const float* c_bo = (const float*)d_in[10];
  const float* e_wq = (const float*)d_in[11];
  const float* e_bq = (const float*)d_in[12];
  const float* e_aq = (const float*)d_in[13];
  const float* e_gq = (const float*)d_in[14];
  const float* e_wk = (const float*)d_in[15];
  const float* e_bk = (const float*)d_in[16];
  const float* e_ak = (const float*)d_in[17];
  const float* e_gk = (const float*)d_in[18];
  const float* e_wv = (const float*)d_in[19];
  const float* e_bv = (const float*)d_in[20];
  const float* e_av = (const float*)d_in[21];
  const float* e_gv = (const float*)d_in[22];
  const float* e_wo = (const float*)d_in[23];
  const float* e_bo = (const float*)d_in[24];
  const float* mask = (const float*)d_in[25];

  const size_t HH  = (size_t)H_ * H_;        // 589824
  const size_t TOK = (size_t)B_ * S_ * H_;   // 12582912

  char* ws = (char*)d_ws;
  size_t off = 0;
  auto alloc = [&](size_t bytes) -> char* {
    char* p = ws + off;
    off = (off + bytes + 255) & ~(size_t)255;
    return p;
  };
  int*      routes = (int*)alloc(B_ * sizeof(int));
  uint16_t* xb     = (uint16_t*)alloc(TOK * 2);
  uint16_t* WQr0 = (uint16_t*)alloc(HH * 2), *WQr1 = (uint16_t*)alloc(HH * 2);
  uint16_t* WKr0 = (uint16_t*)alloc(HH * 2), *WKr1 = (uint16_t*)alloc(HH * 2);
  uint16_t* WVr0 = (uint16_t*)alloc(HH * 2), *WVr1 = (uint16_t*)alloc(HH * 2);
  uint16_t* WOr0 = (uint16_t*)alloc(HH * 2), *WOr1 = (uint16_t*)alloc(HH * 2);
  uint16_t* WQc  = (uint16_t*)alloc(HH * 2);
  uint16_t* WKc  = (uint16_t*)alloc(HH * 2);
  uint16_t* WVc  = (uint16_t*)alloc(HH * 2);
  uint16_t* WOc  = (uint16_t*)alloc(HH * 2);
  uint16_t* Qr   = (uint16_t*)alloc(TOK * 2);
  uint16_t* Kr   = (uint16_t*)alloc(TOK * 2);
  uint16_t* Vrt  = (uint16_t*)alloc(TOK * 2);   // [B][H][S]
  uint16_t* Qc   = (uint16_t*)alloc(TOK * 2);
  uint16_t* Kc   = (uint16_t*)alloc(TOK * 2);
  uint16_t* Vct  = (uint16_t*)alloc(TOK * 2);
  uint16_t* AtR  = (uint16_t*)alloc(TOK * 2);
  uint16_t* AtC  = (uint16_t*)alloc(TOK * 2);

  // 1) routing
  router_kernel<<<B_, 256, 0, stream>>>(x, switch_w, switch_b, routes);

  // 2) x -> bf16
  cvt_bf16_kernel<<<(int)((TOK + 255) / 256), 256, 0, stream>>>(x, xb, (int)TOK);

  // 3) weight prep
  int wblk = (int)((HH + 255) / 256);
  lora_combine_kernel<<<wblk, 256, 0, stream>>>(e_wq + 0 * HH, e_aq + 0 * (size_t)H_ * R_, e_gq + 0 * (size_t)R_ * H_, WQr0);
  lora_combine_kernel<<<wblk, 256, 0, stream>>>(e_wq + 1 * HH, e_aq + 1 * (size_t)H_ * R_, e_gq + 1 * (size_t)R_ * H_, WQr1);
  lora_combine_kernel<<<wblk, 256, 0, stream>>>(e_wk + 0 * HH, e_ak + 0 * (size_t)H_ * R_, e_gk + 0 * (size_t)R_ * H_, WKr0);
  lora_combine_kernel<<<wblk, 256, 0, stream>>>(e_wk + 1 * HH, e_ak + 1 * (size_t)H_ * R_, e_gk + 1 * (size_t)R_ * H_, WKr1);
  lora_combine_kernel<<<wblk, 256, 0, stream>>>(e_wv + 0 * HH, e_av + 0 * (size_t)H_ * R_, e_gv + 0 * (size_t)R_ * H_, WVr0);
  lora_combine_kernel<<<wblk, 256, 0, stream>>>(e_wv + 1 * HH, e_av + 1 * (size_t)H_ * R_, e_gv + 1 * (size_t)R_ * H_, WVr1);
  tconv_kernel<<<wblk, 256, 0, stream>>>(e_wo + 0 * HH, WOr0);
  tconv_kernel<<<wblk, 256, 0, stream>>>(e_wo + 1 * HH, WOr1);
  tconv_kernel<<<wblk, 256, 0, stream>>>(c_wq, WQc);
  tconv_kernel<<<wblk, 256, 0, stream>>>(c_wk, WKc);
  tconv_kernel<<<wblk, 256, 0, stream>>>(c_wv, WVc);
  tconv_kernel<<<wblk, 256, 0, stream>>>(c_wo, WOc);

  // 4) projections (WMMA + TDM A-tile staging, 32 rows per block)
  dim3 pg(S_ / 32, H_ / 256, B_), pb(128);
  proj_kernel<<<pg, pb, 0, stream>>>(xb, WQr0, WQr1, e_bq, e_bq + H_, routes, 1, Qr, 0);
  proj_kernel<<<pg, pb, 0, stream>>>(xb, WKr0, WKr1, e_bk, e_bk + H_, routes, 1, Kr, 0);
  proj_kernel<<<pg, pb, 0, stream>>>(xb, WVr0, WVr1, e_bv, e_bv + H_, routes, 1, Vrt, 1);
  proj_kernel<<<pg, pb, 0, stream>>>(xb, WQc, WQc, c_bq, c_bq, routes, 0, Qc, 0);
  proj_kernel<<<pg, pb, 0, stream>>>(xb, WKc, WKc, c_bk, c_bk, routes, 0, Kc, 0);
  proj_kernel<<<pg, pb, 0, stream>>>(xb, WVc, WVc, c_bv, c_bv, routes, 0, Vct, 1);

  // 5) attention (WMMA flash-style, K/V heads staged via TDM), routed + common
  dim3 ag(S_ / 64, NH_, B_);
  attn_kernel<<<ag, pb, 0, stream>>>(Qr, Kr, Vrt, mask, AtR);
  attn_kernel<<<ag, pb, 0, stream>>>(Qc, Kc, Vct, mask, AtC);

  // 6) fused output projection + combine (forward scaling == 1)
  outproj_kernel<<<pg, pb, 0, stream>>>(AtR, AtC, WOr0, WOr1, WOc,
                                        e_bo, e_bo + H_, c_bo, routes, (float*)d_out);
  (void)in_sizes; (void)n_in; (void)out_size; (void)ws_size;
}
